// SelectiveSSM_43739946942605
// MI455X (gfx1250) — matrix operand order repaired and checked by reading, not compiled, not run on hardware
//
#include <hip/hip_runtime.h>
#include <hip/hip_bf16.h>
#include <math.h>

// ---- problem dims (fixed by reference) ----
#define D_MODEL 1024
#define D_STATE 16
#define D_CONVK 4
#define D_INNER 2048
#define BATCH   2
#define SEQ     2048
#define M_TOTAL (BATCH * SEQ)          // 4096 rows (b,t)

#define LDS_STRIDE 40                  // 32 f16 payload + 8 pad (80B = 20 banks)

typedef __attribute__((ext_vector_type(16))) _Float16 v16h;
typedef __attribute__((ext_vector_type(8)))  _Float16 v8h;
typedef __attribute__((ext_vector_type(8)))  float    v8f;

typedef unsigned int u32x4 __attribute__((ext_vector_type(4)));
typedef int          i32x8 __attribute__((ext_vector_type(8)));
typedef int          i32x4 __attribute__((ext_vector_type(4)));

// ---------------------------------------------------------------------------
// Async global->LDS 16B copy (gfx1250 VGLOBAL async op, tracked by ASYNCcnt)
__device__ __forceinline__ void async_cp16(const _Float16* g, _Float16* l) {
  unsigned lds = (unsigned)(size_t)l;               // low 32 bits = LDS offset
  unsigned long long ga = (unsigned long long)(size_t)g;
  asm volatile("global_load_async_to_lds_b128 %0, %1, off"
               :: "v"(lds), "v"(ga) : "memory");
}
__device__ __forceinline__ void wait_async0() {
  asm volatile("s_wait_asynccnt 0" ::: "memory");
}

// Stage a 128x32 f16 tile (row-major src, leading dim ldk) into LDS
// (row stride LDS_STRIDE). 512 chunks of 16B, 2 per thread, row-contiguous
// across groups of 4 threads for coalescing.
__device__ __forceinline__ void stage_tile(const _Float16* __restrict__ src, int ldk,
                                           int rowBase, int kb,
                                           _Float16* dst, int tid) {
#pragma unroll
  for (int u = 0; u < 2; ++u) {
    int c = tid + u * 256;                  // 0..511
    int r = c >> 2;
    int cc = (c & 3) * 8;
    async_cp16(src + (size_t)(rowBase + r) * ldk + kb + cc,
               dst + r * LDS_STRIDE + cc);
  }
}

// ---------------------------------------------------------------------------
// WMMA fragment loaders from LDS tiles (CDNA5 ISA 7.12.2 layouts).
// A frag: lane l -> M = l%16; elems 0..7 = K half*8+0..7, 8..15 = K 16+half*8+0..7
__device__ __forceinline__ v16h lds_fragA(const _Float16* base, int lane) {
  const _Float16* p = base + (lane & 15) * LDS_STRIDE + ((lane >> 4) << 3);
  v8h lo = *(const v8h*)p;
  v8h hi = *(const v8h*)(p + 16);
  v16h f;
#pragma unroll
  for (int i = 0; i < 8; ++i) { f[i] = lo[i]; f[i + 8] = hi[i]; }
  return f;
}
// B frag (from N-major tile): lane l -> N = l%16; elems 0..7 = K half*8+0..7, 8..15 = K 16+half*8+0..7
__device__ __forceinline__ v16h lds_fragB(const _Float16* base, int lane) {
  const _Float16* p = base + (lane & 15) * LDS_STRIDE + ((lane >> 4) << 3);
  v8h lo = *(const v8h*)p;
  v8h hi = *(const v8h*)(p + 16);
  v16h f;
#pragma unroll
  for (int i = 0; i < 8; ++i) { f[i] = lo[i]; f[i + 8] = hi[i]; }
  return f;
}

// Direct-from-global fragment loaders (used by the narrow Bp GEMM)
__device__ __forceinline__ v16h load_fragA(const _Float16* __restrict__ A,
                                           int ldk, int mbase, int kb, int lane) {
  const _Float16* p = A + (size_t)(mbase + (lane & 15)) * ldk + kb + ((lane >> 4) << 3);
  v8h lo = *(const v8h*)p;
  v8h hi = *(const v8h*)(p + 16);
  v16h f;
#pragma unroll
  for (int i = 0; i < 8; ++i) { f[i] = lo[i]; f[i + 8] = hi[i]; }
  return f;
}
__device__ __forceinline__ v16h load_fragB(const _Float16* __restrict__ BT,
                                           int ldk, int nbase, int kb, int lane) {
  const _Float16* p = BT + (size_t)(nbase + (lane & 15)) * ldk + kb + ((lane >> 4) << 3);
  v8h lo = *(const v8h*)p;
  v8h hi = *(const v8h*)(p + 16);
  v16h f;
#pragma unroll
  for (int i = 0; i < 8; ++i) { f[i] = lo[i]; f[i + 8] = hi[i]; }
  return f;
}

__device__ __forceinline__ v8f wmma16(v16h a, v16h b, v8f c) {
  return __builtin_amdgcn_wmma_f32_16x16x32_f16(false, a, false, b, (short)0, c,
                                                false, false);
}

// ---------------------------------------------------------------------------
// TDM: single-descriptor 1-D copy of nElem fp32 from global to LDS.
// D# per CDNA5 ISA 8.3/8.4: group0 = {count|flags, lds_addr, global_addr, type=2},
// group1 = {data_size=4B, tensor_dim0=tile_dim0=nElem, tensor_dim1=tile_dim1=1}.
// This toolchain exposes the 6-arg builtin: (g0, g1, g2, g3, g?, cpol).
__device__ __forceinline__ void tdm_load_f32(const float* gsrc, float* lds, unsigned nElem) {
  unsigned long long ga = (unsigned long long)(size_t)gsrc;
  u32x4 g0;
  g0[0] = 1u;                                          // count=1, user descriptor
  g0[1] = (unsigned)(size_t)lds;                       // lds_addr
  g0[2] = (unsigned)ga;                                // global_addr[31:0]
  g0[3] = (unsigned)((ga >> 32) & 0x01ffffffu) | (2u << 30); // addr[56:32] | type=2
  i32x8 g1;
  g1[0] = (int)(2u << 16);                             // data_size = 4 bytes
  g1[1] = (int)((nElem & 0xffffu) << 16);              // tensor_dim0[15:0]
  g1[2] = (int)((nElem >> 16) & 0xffffu) | (1 << 16);  // tensor_dim0[31:16] | tensor_dim1=1
  g1[3] = (int)((nElem & 0xffffu) << 16);              // tile_dim0
  g1[4] = 1;                                           // tile_dim1=1, tile_dim2=0
  g1[5] = (int)nElem;                                  // tensor_dim0_stride[31:0]
  g1[6] = (int)((nElem & 0xffffu) << 16);              // stride0[47:32]=0 | stride1[15:0]
  g1[7] = 0;
  i32x4 zz4 = (i32x4){0, 0, 0, 0};
  i32x8 zz8 = (i32x8){0, 0, 0, 0, 0, 0, 0, 0};
  __builtin_amdgcn_tensor_load_to_lds(g0, g1, zz4, zz4, zz8, 0);
}

// ---------------------------------------------------------------------------
// Tiled WMMA GEMM with async double-buffered LDS staging.
// C(MxN) = A(MxK) * BT(NxK)^T, fp32 accumulate. Block 256 thr, tile 128x128.
// EPI: 0 = split xz, 1 = softplus(acc + bias) -> delta, 2 = plain store
template <int EPI>
__global__ __launch_bounds__(256) void gemm_wmma(
    const _Float16* __restrict__ A, const _Float16* __restrict__ BT,
    int K, int N,
    float* __restrict__ out0, float* __restrict__ out1,
    const float* __restrict__ bias) {
  __shared__ _Float16 sA[2][128 * LDS_STRIDE];
  __shared__ _Float16 sB[2][128 * LDS_STRIDE];

  const int tid = threadIdx.x;
  const int lane = tid & 31;
  const int wave = tid >> 5;
  const int wm = wave >> 1;                 // 0..3
  const int wn = wave & 1;                  // 0..1
  const int mblk = blockIdx.y * 128;
  const int nblk = blockIdx.x * 128;

  v8f acc[2][4];
#pragma unroll
  for (int i = 0; i < 2; ++i)
#pragma unroll
    for (int j = 0; j < 4; ++j) acc[i][j] = (v8f){0.f, 0.f, 0.f, 0.f, 0.f, 0.f, 0.f, 0.f};

  stage_tile(A, K, mblk, 0, sA[0], tid);
  stage_tile(BT, K, nblk, 0, sB[0], tid);

  for (int kb = 0; kb < K; kb += 32) {
    const int p = (kb >> 5) & 1;
    wait_async0();            // our staged chunks are in LDS
    __syncthreads();          // everyone's chunks visible; prev compute done
    if (kb + 32 < K) {        // overlap next-tile staging with this tile's math
      stage_tile(A, K, mblk, kb + 32, sA[1 - p], tid);
      stage_tile(BT, K, nblk, kb + 32, sB[1 - p], tid);
    }
    const _Float16* aB = sA[p] + (wm * 32) * LDS_STRIDE;
    const _Float16* bB = sB[p] + (wn * 64) * LDS_STRIDE;
    v16h a0 = lds_fragA(aB, lane);
    v16h a1 = lds_fragA(aB + 16 * LDS_STRIDE, lane);
    v16h b0 = lds_fragB(bB, lane);
    v16h b1 = lds_fragB(bB + 16 * LDS_STRIDE, lane);
    v16h b2 = lds_fragB(bB + 32 * LDS_STRIDE, lane);
    v16h b3 = lds_fragB(bB + 48 * LDS_STRIDE, lane);
    acc[0][0] = wmma16(a0, b0, acc[0][0]);
    acc[0][1] = wmma16(a0, b1, acc[0][1]);
    acc[0][2] = wmma16(a0, b2, acc[0][2]);
    acc[0][3] = wmma16(a0, b3, acc[0][3]);
    acc[1][0] = wmma16(a1, b0, acc[1][0]);
    acc[1][1] = wmma16(a1, b1, acc[1][1]);
    acc[1][2] = wmma16(a1, b2, acc[1][2]);
    acc[1][3] = wmma16(a1, b3, acc[1][3]);
  }

#pragma unroll
  for (int i = 0; i < 2; ++i) {
#pragma unroll
    for (int j = 0; j < 4; ++j) {
      const int row0 = mblk + wm * 32 + i * 16 + ((lane >> 4) << 3);
      const int col = nblk + wn * 64 + j * 16 + (lane & 15);
#pragma unroll
      for (int v = 0; v < 8; ++v) {
        const int row = row0 + v;
        float val = acc[i][j][v];
        if (EPI == 0) {
          if (col < D_INNER) out0[(size_t)row * D_INNER + col] = val;
          else               out1[(size_t)row * D_INNER + (col - D_INNER)] = val;
        } else if (EPI == 1) {
          float t = val + bias[col];
          float sp = (t > 20.f) ? t : log1pf(__expf(t));
          out0[(size_t)row * N + col] = sp;
        } else {
          out0[(size_t)row * N + col] = val;
        }
      }
    }
  }
}

// Narrow GEMM for Bp = xs @ W_x[:,16:32]  (M x 16). One wave per 16-row tile.
__global__ __launch_bounds__(256) void gemm_bproj(
    const _Float16* __restrict__ A, const _Float16* __restrict__ BT /*16 x K*/,
    int K, float* __restrict__ Bp) {
  const int lane = threadIdx.x & 31;
  const int wave = threadIdx.x >> 5;
  const int m = (blockIdx.x * 8 + wave) * 16;
  v8f acc = (v8f){0.f, 0.f, 0.f, 0.f, 0.f, 0.f, 0.f, 0.f};
  for (int kb = 0; kb < K; kb += 32) {
    v16h a = load_fragA(A, K, m, kb, lane);
    v16h b = load_fragB(BT, K, 0, kb, lane);
    acc = wmma16(a, b, acc);
  }
  const int row0 = m + ((lane >> 4) << 3);
  const int col = lane & 15;
#pragma unroll
  for (int v = 0; v < 8; ++v) Bp[(size_t)(row0 + v) * D_STATE + col] = acc[v];
}

// ---------------------------------------------------------------------------
// Weight prep: fp32 (R x C) --transpose+cvt--> f16 (numC x R), column window.
__global__ void transpose_to_h(const float* __restrict__ src, _Float16* __restrict__ dst,
                               int R, int C, int colOff, int numC) {
  size_t idx = (size_t)blockIdx.x * blockDim.x + threadIdx.x;
  if (idx >= (size_t)R * numC) return;
  int c = (int)(idx / R);
  int r = (int)(idx % R);
  dst[idx] = (_Float16)src[(size_t)r * C + colOff + c];
}

__global__ void cvt_to_h(const float* __restrict__ s, _Float16* __restrict__ d, size_t n) {
  size_t i = (size_t)blockIdx.x * blockDim.x + threadIdx.x;
  if (i < n) d[i] = (_Float16)s[i];
}

// ---------------------------------------------------------------------------
// Depthwise causal conv (K=4) + bias + silu; fp32 (for scan) + f16 (for GEMMs)
__global__ __launch_bounds__(256) void conv_silu(
    const float* __restrict__ xs_pre, const float* __restrict__ cw,
    const float* __restrict__ cb, float* __restrict__ xs, _Float16* __restrict__ xs_h) {
  size_t i = (size_t)blockIdx.x * 256 + threadIdx.x;
  const size_t total = (size_t)M_TOTAL * D_INNER;
  if (i >= total) return;
  int d = (int)(i % D_INNER);
  size_t bt = i / D_INNER;
  int t = (int)(bt % SEQ);
  int b = (int)(bt / SEQ);
  float acc = cb[d];
#pragma unroll
  for (int k = 0; k < D_CONVK; ++k) {
    int tt = t + k - (D_CONVK - 1);
    if (tt >= 0) acc += cw[d * D_CONVK + k] * xs_pre[((size_t)b * SEQ + tt) * D_INNER + d];
  }
  float s = acc / (1.f + __expf(-acc));
  xs[i] = s;
  xs_h[i] = (_Float16)s;
}

// ---------------------------------------------------------------------------
// Selective scan. One lane per channel; 16-state recurrence in registers.
// Bp for the whole batch sequence (128 KB) staged in LDS by the TDM.
__global__ __launch_bounds__(256) void scan_ssm(
    const float* __restrict__ delta, const float* __restrict__ xs,
    const float* __restrict__ z, const float* __restrict__ Bp,
    const float* __restrict__ A_log, const float* __restrict__ D_param,
    _Float16* __restrict__ yact) {
  extern __shared__ float sB[];                       // SEQ*16 floats = 128 KB
  const int b = blockIdx.y;
  const int ch = blockIdx.x * 256 + threadIdx.x;

  if (threadIdx.x == 0)
    tdm_load_f32(Bp + (size_t)b * SEQ * D_STATE, sB, SEQ * D_STATE);
  __builtin_amdgcn_s_wait_tensorcnt(0);
  __syncthreads();

  float Av[D_STATE];
#pragma unroll
  for (int j = 0; j < D_STATE; ++j) Av[j] = -__expf(A_log[(size_t)ch * D_STATE + j]);
  const float Dp = D_param[ch];

  float h[D_STATE];
#pragma unroll
  for (int j = 0; j < D_STATE; ++j) h[j] = 0.f;

  const float4* sB4 = (const float4*)sB;
  const size_t base = (size_t)b * SEQ * D_INNER + ch;
  for (int t = 0; t < SEQ; ++t) {
    const size_t off = base + (size_t)t * D_INNER;
    if (t + 8 < SEQ) {
      __builtin_prefetch(&delta[off + 8 * D_INNER]);
      __builtin_prefetch(&xs[off + 8 * D_INNER]);
      __builtin_prefetch(&z[off + 8 * D_INNER]);
    }
    const float dlt = delta[off];
    const float xv = xs[off];
    const float zv = z[off];
    const float du = dlt * xv;

    float4 B0 = sB4[t * 4 + 0];
    float4 B1 = sB4[t * 4 + 1];
    float4 B2 = sB4[t * 4 + 2];
    float4 B3 = sB4[t * 4 + 3];
    float Bt[D_STATE] = {B0.x, B0.y, B0.z, B0.w, B1.x, B1.y, B1.z, B1.w,
                         B2.x, B2.y, B2.z, B2.w, B3.x, B3.y, B3.z, B3.w};
    float accsum = 0.f;
#pragma unroll
    for (int j = 0; j < D_STATE; ++j) {
      float dA = __expf(dlt * Av[j]);
      h[j] = dA * h[j] + du * Bt[j];
      accsum += h[j];
    }
    float y = accsum + xv * Dp;
    float zs = zv / (1.f + __expf(-zv));
    yact[off] = (_Float16)(y * zs);
  }
}

// ---------------------------------------------------------------------------
extern "C" void kernel_launch(void* const* d_in, const int* in_sizes, int n_in,
                              void* d_out, int out_size, void* d_ws, size_t ws_size,
                              hipStream_t stream) {
  const float* x       = (const float*)d_in[0];  // (2,2048,1024)
  const float* W_in    = (const float*)d_in[1];  // (1024,4096)
  const float* conv_w  = (const float*)d_in[2];  // (2048,4)
  const float* conv_b  = (const float*)d_in[3];  // (2048,)
  const float* W_x     = (const float*)d_in[4];  // (2048,32)
  const float* W_dt    = (const float*)d_in[5];  // (2048,2048)
  const float* b_dt    = (const float*)d_in[6];  // (2048,)
  const float* A_log   = (const float*)d_in[7];  // (2048,16)
  const float* D_param = (const float*)d_in[8];  // (2048,)
  const float* W_out   = (const float*)d_in[9];  // (2048,1024)
  float* out = (float*)d_out;                    // (2,2048,1024)

  char* w = (char*)d_ws;
  size_t off = 0;
  auto alloc = [&](size_t bytes) {
    void* p = w + off;
    off += (bytes + 255) & ~(size_t)255;
    return p;
  };
  _Float16* xh     = (_Float16*)alloc((size_t)M_TOTAL * D_MODEL * 2);
  _Float16* WinT   = (_Float16*)alloc((size_t)(2 * D_INNER) * D_MODEL * 2);
  _Float16* WdtT   = (_Float16*)alloc((size_t)D_INNER * D_INNER * 2);
  _Float16* WxBT   = (_Float16*)alloc((size_t)D_STATE * D_INNER * 2);
  _Float16* WoutT  = (_Float16*)alloc((size_t)D_MODEL * D_INNER * 2);
  float*    xs_pre = (float*)alloc((size_t)M_TOTAL * D_INNER * 4);
  float*    zbuf   = (float*)alloc((size_t)M_TOTAL * D_INNER * 4);
  float*    xsbuf  = (float*)alloc((size_t)M_TOTAL * D_INNER * 4);
  _Float16* xsh    = (_Float16*)alloc((size_t)M_TOTAL * D_INNER * 2);
  float*    dbuf   = (float*)alloc((size_t)M_TOTAL * D_INNER * 4);
  float*    Bpbuf  = (float*)alloc((size_t)M_TOTAL * D_STATE * 4);
  _Float16* yacth  = (_Float16*)alloc((size_t)M_TOTAL * D_INNER * 2);
  (void)ws_size; (void)in_sizes; (void)n_in; (void)out_size;

  const int TB = 256;
  auto blocks = [](size_t n, int tb) { return (unsigned)((n + tb - 1) / tb); };

  // 1) convert / transpose weights + activations to f16
  cvt_to_h<<<blocks((size_t)M_TOTAL * D_MODEL, TB), TB, 0, stream>>>(
      x, xh, (size_t)M_TOTAL * D_MODEL);
  transpose_to_h<<<blocks((size_t)D_MODEL * 2 * D_INNER, TB), TB, 0, stream>>>(
      W_in, WinT, D_MODEL, 2 * D_INNER, 0, 2 * D_INNER);
  transpose_to_h<<<blocks((size_t)D_INNER * D_INNER, TB), TB, 0, stream>>>(
      W_dt, WdtT, D_INNER, D_INNER, 0, D_INNER);
  transpose_to_h<<<blocks((size_t)D_INNER * D_STATE, TB), TB, 0, stream>>>(
      W_x, WxBT, D_INNER, 2 * D_STATE, D_STATE, D_STATE);
  transpose_to_h<<<blocks((size_t)D_INNER * D_MODEL, TB), TB, 0, stream>>>(
      W_out, WoutT, D_INNER, D_MODEL, 0, D_MODEL);

  // 2) xz = x @ W_in  -> xs_pre | z
  gemm_wmma<0><<<dim3((2 * D_INNER) / 128, M_TOTAL / 128), TB, 0, stream>>>(
      xh, WinT, D_MODEL, 2 * D_INNER, xs_pre, zbuf, nullptr);

  // 3) depthwise causal conv + silu
  conv_silu<<<blocks((size_t)M_TOTAL * D_INNER, TB), TB, 0, stream>>>(
      xs_pre, conv_w, conv_b, xsbuf, xsh);

  // 4) delta = softplus(xs @ W_dt + b_dt)
  gemm_wmma<1><<<dim3(D_INNER / 128, M_TOTAL / 128), TB, 0, stream>>>(
      xsh, WdtT, D_INNER, D_INNER, dbuf, nullptr, b_dt);

  // 5) Bp = xs @ W_x[:, 16:32]
  gemm_bproj<<<M_TOTAL / 128, TB, 0, stream>>>(xsh, WxBT, D_INNER, Bpbuf);

  // 6) selective scan + output gating (TDM stages Bp into LDS)
  scan_ssm<<<dim3(D_INNER / 256, BATCH), TB, SEQ * D_STATE * sizeof(float), stream>>>(
      dbuf, xsbuf, zbuf, Bpbuf, A_log, D_param, yacth);

  // 7) out = yact @ W_out
  gemm_wmma<2><<<dim3(D_MODEL / 128, M_TOTAL / 128), TB, 0, stream>>>(
      yacth, WoutT, D_INNER, D_MODEL, out, nullptr, nullptr);
}
